// FeedForwardQuantum_65481071405447
// MI455X (gfx1250) — compile-verified, matrix operand order checked by reading, not run on hardware
//
#include <hip/hip_runtime.h>
#include <hip/hip_bf16.h>

// ---------------------------------------------------------------------------
// FeedForwardQuantum for MI455X (gfx1250, wave32, WMMA)
//   x[16384,768] -> q[16384,10] -> h[16384,3072] -> out[16384,768]
//   Stage 3 (77 GFLOP) on v_wmma_f32_16x16x32_bf16 with LDS-staged,
//   double-buffered 128x256 workgroup tiles (64x64 per wave).
// Workspace layout (d_ws):
//   [0)        W2 as bf16, layout [768][3072]   (4,718,592 B)
//   [4718592)  h  as bf16, layout [16384][3072] (100,663,296 B, L2-resident)
// ---------------------------------------------------------------------------

typedef __attribute__((ext_vector_type(16))) __bf16 v16bf;
typedef __attribute__((ext_vector_type(8)))  __bf16 v8bf;
typedef __attribute__((ext_vector_type(4)))  __bf16 v4bf;
typedef __attribute__((ext_vector_type(8)))  float  v8f;

#define EMBED 768
#define FFN   3072
#define NW    10
#define ROWS  16384          // 8 * 2048

// LDS tile geometry: rows of 32 bf16 (64 B) padded to 80 B stride.
// 20*n mod 64 is injective for n=0..15 -> conflict-free ds_load_b128 frags.
#define ROWB      80
#define A_ROWS    128
#define B_ROWS    256
#define A_BYTES   (A_ROWS * ROWB)            // 10240
#define B_BYTES   (B_ROWS * ROWB)            // 20480
#define STAGE_B   (A_BYTES + B_BYTES)        // 30720 per buffer
#define KSTEPS    (FFN / 32)                 // 96

// ---------------------------------------------------------------------------
// Kernel 1: convert W2 (f32 [768][3072]) -> bf16, same layout.
// ---------------------------------------------------------------------------
__global__ __launch_bounds__(256) void qffn_cvt_w2(
    const float* __restrict__ src, __bf16* __restrict__ dst, int n4)
{
    int i = blockIdx.x * blockDim.x + threadIdx.x;
    if (i >= n4) return;
    float4 v = ((const float4*)src)[i];
    v4bf o;
    o[0] = (__bf16)v.x; o[1] = (__bf16)v.y;
    o[2] = (__bf16)v.z; o[3] = (__bf16)v.w;
    ((v4bf*)dst)[i] = o;
}

// ---------------------------------------------------------------------------
// Kernel 2: per wave, two rows:
//   q_w = cos(theta_w) * cos( dot(x_row, Wmap_w) ),  w = 0..9
//   h_f = relu( b1_f + sum_w q_w * W1[f][w] )  -> bf16
// ---------------------------------------------------------------------------
__global__ __launch_bounds__(256) void qffn_qh(
    const float* __restrict__ x,      // [ROWS][EMBED]
    const float* __restrict__ Wm,     // [NW][EMBED]
    const float* __restrict__ theta,  // [NW]
    const float* __restrict__ W1,     // [FFN][NW]
    const float* __restrict__ b1,     // [FFN]
    __bf16* __restrict__ H)           // [ROWS][FFN] bf16 out
{
    const int lane = threadIdx.x & 31;
    const int wave = threadIdx.x >> 5;
    const int wid  = blockIdx.x * 8 + wave;   // 0..8191
    const int r0   = wid * 2;                 // two rows per wave

    float p0[NW], p1[NW];
#pragma unroll
    for (int w = 0; w < NW; ++w) { p0[w] = 0.f; p1[w] = 0.f; }

    const float* x0 = x + (size_t)r0 * EMBED + lane * 24;
    const float* x1 = x0 + EMBED;
#pragma unroll
    for (int c = 0; c < 6; ++c) {
        float4 xa = *(const float4*)(x0 + c * 4);
        float4 xb = *(const float4*)(x1 + c * 4);
#pragma unroll
        for (int w = 0; w < NW; ++w) {
            float4 m = *(const float4*)(Wm + w * EMBED + lane * 24 + c * 4);
            p0[w] = fmaf(xa.x, m.x, fmaf(xa.y, m.y, fmaf(xa.z, m.z, fmaf(xa.w, m.w, p0[w]))));
            p1[w] = fmaf(xb.x, m.x, fmaf(xb.y, m.y, fmaf(xb.z, m.z, fmaf(xb.w, m.w, p1[w]))));
        }
    }

    float q0[NW], q1[NW];
#pragma unroll
    for (int w = 0; w < NW; ++w) {
        float s0 = p0[w], s1 = p1[w];
#pragma unroll
        for (int m = 16; m >= 1; m >>= 1) {
            s0 += __shfl_xor(s0, m, 32);
            s1 += __shfl_xor(s1, m, 32);
        }
        float ct = __builtin_cosf(theta[w]);
        q0[w] = ct * __builtin_cosf(s0);
        q1[w] = ct * __builtin_cosf(s1);
    }

    __bf16* h0 = H + (size_t)r0 * FFN;
    __bf16* h1 = h0 + FFN;
#pragma unroll 4
    for (int t = 0; t < FFN / 32; ++t) {
        const int f = t * 32 + lane;          // coalesced across the wave
        const float* w1r = W1 + (size_t)f * NW;
        float a0 = b1[f], a1 = a0;
#pragma unroll
        for (int w = 0; w < NW; w += 2) {     // 8-byte aligned float2 loads
            float2 ww = *(const float2*)(w1r + w);
            a0 = fmaf(q0[w], ww.x, a0); a0 = fmaf(q0[w + 1], ww.y, a0);
            a1 = fmaf(q1[w], ww.x, a1); a1 = fmaf(q1[w + 1], ww.y, a1);
        }
        h0[f] = (__bf16)fmaxf(a0, 0.f);
        h1[f] = (__bf16)fmaxf(a1, 0.f);
    }
}

// ---------------------------------------------------------------------------
// Kernel 3: out = h @ W2^T + b2 on v_wmma_f32_16x16x32_bf16.
// Block = 256 threads = 8 waves in a 2(M) x 4(N) grid of 64x64 wave tiles.
// WG tile 128x256; k-chunks of 32 staged into double-buffered LDS.
// 16-bit fragment layout (ISA 7.12.2): lane<16 -> K {k..k+7, k+16..k+23},
// lane>=16 -> K {k+8..k+15, k+24..k+31}; each half is a 16 B ds_load_b128.
// ---------------------------------------------------------------------------
__device__ __forceinline__ v16bf lds_frag(const unsigned char* p)
{
    v8bf lo = *(const v8bf*)p;          // K = khalf .. khalf+7
    v8bf hi = *(const v8bf*)(p + 32);   // K = khalf+16 .. khalf+23
    v16bf r;
#pragma unroll
    for (int i = 0; i < 8; ++i) { r[i] = lo[i]; r[8 + i] = hi[i]; }
    return r;
}

__global__ __launch_bounds__(256) void qffn_gemm(
    const __bf16* __restrict__ H,    // [ROWS][FFN] bf16
    const __bf16* __restrict__ Wb,   // [EMBED][FFN] bf16 (W2)
    const float*  __restrict__ b2,   // [EMBED]
    float* __restrict__ out)         // [ROWS][EMBED]
{
    __shared__ __align__(16) unsigned char smem[2 * STAGE_B];

    const int tid    = threadIdx.x;
    const int lane   = tid & 31;
    const int wave   = tid >> 5;
    const int l16    = lane & 15;
    const int khalf  = (lane < 16) ? 0 : 8;    // fragment k split per ISA layout
    const int wave_m = wave >> 2;              // 0..1
    const int wave_n = wave & 3;               // 0..3

    const int rowA0 = blockIdx.x * 128;        // WG M origin
    const int n0    = blockIdx.y * 256;        // WG N origin

    // ---- cooperative staging addresses: 16 B chunk per (row-quarter) ------
    // A: 128 rows x 4 chunks = 512 chunks -> thread does c = tid, tid+256
    // B: 256 rows x 4 chunks = 1024 chunks -> thread does c = tid + 256*i
    const int srow = tid >> 2;                 // 0..63
    const int sq   = tid & 3;                  // 16 B chunk within 64 B row
    const char* gA0 = (const char*)(H  + (size_t)(rowA0 + srow) * FFN + sq * 8);
    const char* gA1 = gA0 + (size_t)64 * FFN * 2;
    const char* gB0 = (const char*)(Wb + (size_t)(n0 + srow) * FFN + sq * 8);
    const char* gB1 = gB0 + (size_t)64  * FFN * 2;
    const char* gB2 = gB0 + (size_t)128 * FFN * 2;
    const char* gB3 = gB0 + (size_t)192 * FFN * 2;
    const int ldA0 = srow * ROWB + sq * 16;
    const int ldA1 = ldA0 + 64 * ROWB;
    const int ldB0 = A_BYTES + srow * ROWB + sq * 16;
    const int ldB1 = ldB0 + 64  * ROWB;
    const int ldB2 = ldB0 + 128 * ROWB;
    const int ldB3 = ldB0 + 192 * ROWB;

    // ---- fragment read offsets within a stage buffer ----------------------
    const int fragA = (wave_m * 64 + l16) * ROWB + khalf * 2;            // + mb*16*ROWB
    const int fragB = A_BYTES + (wave_n * 64 + l16) * ROWB + khalf * 2;  // + nb*16*ROWB

    v8f zero = {};
    v8f acc[4][4];
#pragma unroll
    for (int mb = 0; mb < 4; ++mb)
#pragma unroll
        for (int nb = 0; nb < 4; ++nb) acc[mb][nb] = zero;

    // ---- prologue: stage k-chunk 0 into buffer 0 --------------------------
    {
        uint4 a0 = *(const uint4*)gA0, a1 = *(const uint4*)gA1;
        uint4 c0 = *(const uint4*)gB0, c1 = *(const uint4*)gB1;
        uint4 c2 = *(const uint4*)gB2, c3 = *(const uint4*)gB3;
        *(uint4*)(smem + ldA0) = a0; *(uint4*)(smem + ldA1) = a1;
        *(uint4*)(smem + ldB0) = c0; *(uint4*)(smem + ldB1) = c1;
        *(uint4*)(smem + ldB2) = c2; *(uint4*)(smem + ldB3) = c3;
    }
    __syncthreads();

    int cur = 0;
    for (int ks = 0; ks < KSTEPS; ++ks) {
        // issue global loads for the next k-chunk while computing this one
        uint4 a0, a1, c0, c1, c2, c3;
        const bool more = (ks + 1) < KSTEPS;
        if (more) {
            const size_t kb = (size_t)(ks + 1) * 64;   // 32 bf16 = 64 B per row
            a0 = *(const uint4*)(gA0 + kb); a1 = *(const uint4*)(gA1 + kb);
            c0 = *(const uint4*)(gB0 + kb); c1 = *(const uint4*)(gB1 + kb);
            c2 = *(const uint4*)(gB2 + kb); c3 = *(const uint4*)(gB3 + kb);
        }

        // fragments from LDS, 16 WMMAs
        const unsigned char* buf = smem + cur * STAGE_B;
        v16bf afr[4], bfr[4];
#pragma unroll
        for (int mb = 0; mb < 4; ++mb) afr[mb] = lds_frag(buf + fragA + mb * 16 * ROWB);
#pragma unroll
        for (int nb = 0; nb < 4; ++nb) bfr[nb] = lds_frag(buf + fragB + nb * 16 * ROWB);
#pragma unroll
        for (int mb = 0; mb < 4; ++mb)
#pragma unroll
            for (int nb = 0; nb < 4; ++nb)
                acc[mb][nb] = __builtin_amdgcn_wmma_f32_16x16x32_bf16(
                    false, afr[mb], false, bfr[nb], (short)0, acc[mb][nb],
                    false, false);

        // store staged chunk into the other buffer, then rotate
        if (more) {
            unsigned char* nb_ = smem + (cur ^ 1) * STAGE_B;
            *(uint4*)(nb_ + ldA0) = a0; *(uint4*)(nb_ + ldA1) = a1;
            *(uint4*)(nb_ + ldB0) = c0; *(uint4*)(nb_ + ldB1) = c1;
            *(uint4*)(nb_ + ldB2) = c2; *(uint4*)(nb_ + ldB3) = c3;
        }
        __syncthreads();
        cur ^= 1;
    }

    // ---- epilogue ---------------------------------------------------------
    // C/D layout: VGPR r -> M = r (lanes 0-15) / M = r+8 (lanes 16-31), N = lane&15
    const int rbase = rowA0 + wave_m * 64 + ((lane < 16) ? 0 : 8);
    const int cbase = n0 + wave_n * 64 + l16;
#pragma unroll
    for (int nb = 0; nb < 4; ++nb) {
        const int col = cbase + nb * 16;
        const float bias = b2[col];
#pragma unroll
        for (int mb = 0; mb < 4; ++mb) {
            const int rw = rbase + mb * 16;
#pragma unroll
            for (int r = 0; r < 8; ++r)
                out[(size_t)(rw + r) * EMBED + col] = acc[mb][nb][r] + bias;
        }
    }
}

// ---------------------------------------------------------------------------
extern "C" void kernel_launch(void* const* d_in, const int* in_sizes, int n_in,
                              void* d_out, int out_size, void* d_ws, size_t ws_size,
                              hipStream_t stream)
{
    const float* x     = (const float*)d_in[0];
    const float* Wm    = (const float*)d_in[1];
    const float* theta = (const float*)d_in[2];
    const float* W1    = (const float*)d_in[3];
    const float* b1    = (const float*)d_in[4];
    const float* W2    = (const float*)d_in[5];
    const float* b2    = (const float*)d_in[6];
    float* out = (float*)d_out;

    __bf16* wb = (__bf16*)d_ws;                               // 768*3072 bf16
    __bf16* hb = (__bf16*)((char*)d_ws +
                           (size_t)EMBED * FFN * sizeof(__bf16)); // 16384*3072 bf16

    // K1: W2 f32 -> bf16 (768*3072/4 = 589824 float4 elements)
    qffn_cvt_w2<<<2304, 256, 0, stream>>>(W2, wb, EMBED * FFN / 4);

    // K2: q + h (8 waves/block * 2 rows/wave -> 1024 blocks for 16384 rows)
    qffn_qh<<<1024, 256, 0, stream>>>(x, Wm, theta, W1, b1, hb);

    // K3: WMMA GEMM, grid 128 x 3 tiles of 128x256
    qffn_gemm<<<dim3(ROWS / 128, EMBED / 256), 256, 0, stream>>>(hb, wb, b2, out);
}
